// MLPRNN_70308614636200
// MI455X (gfx1250) — compile-verified, matrix operand order checked by reading, not run on hardware
//
#include <hip/hip_runtime.h>
#include <hip/hip_bf16.h>

// MLP-RNN: per step  h = tanh([x_t, h_prev] @ W1 + b1) @ W2 + b2
// L=512 sequential steps -> 2 dependent WMMA-GEMM kernels per step.
// f16 WMMA (v_wmma_f32_16x16x32_f16) with f32 accumulation; weights converted
// once per call to pre-transposed f16 (B fragments contiguous along K, served
// from the 192MB L2 every step). Double-buffered LDS tiles staged with
// GLOBAL_LOAD_ASYNC_TO_LDS_B128; one barrier per K-step. The pipeline is
// manually unrolled x2 with fixed buffer indices and an even trip count so
// the compiler keeps accumulators pinned (no v_mov churn / hazard NOPs).

#define RNN_L 512
#define RNN_B 256
#define RNN_D 512
#define RNN_H 512
#define RNN_M 2048
#define K1 (RNN_D + RNN_H)   // 1024
#define K2 RNN_M             // 2048

#define LDSK 40              // padded K-stride (halves): conflict-free b128 reads

typedef __attribute__((ext_vector_type(16))) _Float16 v16h;
typedef __attribute__((ext_vector_type(8)))  _Float16 v8h;
typedef __attribute__((ext_vector_type(8)))  float    v8f;
typedef __attribute__((vector_size(16)))     int      v4i_;

#define AS1 __attribute__((address_space(1)))
#define AS3 __attribute__((address_space(3)))

#if __has_builtin(__builtin_amdgcn_global_load_async_to_lds_b128) && \
    __has_builtin(__builtin_amdgcn_s_wait_asynccnt)
#define USE_ASYNC_LDS 1
#else
#define USE_ASYNC_LDS 0
#endif

static __device__ __forceinline__ v16h cat8(v8h lo, v8h hi) {
  return __builtin_shufflevector(lo, hi, 0,1,2,3,4,5,6,7,8,9,10,11,12,13,14,15);
}

static __device__ __forceinline__ v8h cvt8(float4 a, float4 b) {
  v8h r;
  r[0]=(_Float16)a.x; r[1]=(_Float16)a.y; r[2]=(_Float16)a.z; r[3]=(_Float16)a.w;
  r[4]=(_Float16)b.x; r[5]=(_Float16)b.y; r[6]=(_Float16)b.z; r[7]=(_Float16)b.w;
  return r;
}

// 16B global->LDS copy: async DMA when the toolchain exposes it, else VGPR copy.
static __device__ __forceinline__ void copy16B_to_lds(_Float16* dst,
                                                      const _Float16* src) {
#if USE_ASYNC_LDS
  __builtin_amdgcn_global_load_async_to_lds_b128(
      (AS1 v4i_*)(AS1 void*)(void*)src,
      (AS3 v4i_*)(AS3 void*)(void*)dst, 0, 0);
#else
  *(v8h*)dst = *(const v8h*)src;
#endif
}

static __device__ __forceinline__ void async_wait0() {
#if USE_ASYNC_LDS
  __builtin_amdgcn_s_wait_asynccnt(0);
#endif
}

// ---------------------------------------------------------------------------
// One-time (per call): W1 [K1, M] f32 -> W1t [M, K1] f16 ; W2 [K2, H] -> W2t [H, K2]
// ---------------------------------------------------------------------------
__global__ __launch_bounds__(256) void rnn_convert_weights(
    const float* __restrict__ W1, const float* __restrict__ W2,
    _Float16* __restrict__ W1t, _Float16* __restrict__ W2t) {
  int idx = blockIdx.x * 256 + threadIdx.x;
  const int n1 = RNN_M * K1;                 // 2048*1024
  if (idx < n1) {
    int n = idx >> 10, k = idx & 1023;       // W1t[n][k] = W1[k][n]
    W1t[idx] = (_Float16)W1[(size_t)k * RNN_M + n];
  } else {
    int i = idx - n1;
    int h = i >> 11, m = i & 2047;           // W2t[h][m] = W2[m][h]
    W2t[i] = (_Float16)W2[(size_t)m * RNN_H + h];
  }
}

// ---------------------------------------------------------------------------
// GEMM1: act[B, M] = tanh( [x_t | h_prev] @ W1 + b1 )   (A: f32->f16 on the fly)
// Block tile 64x128, BK=32, 8 waves; wave tile 16x64 = 4 WMMA accumulators.
// ---------------------------------------------------------------------------
__global__ __launch_bounds__(256, 1) void rnn_gemm1_tanh(
    const float* __restrict__ x, const float* __restrict__ hprev,
    const _Float16* __restrict__ W1t, const float* __restrict__ b1,
    _Float16* __restrict__ act) {
  __shared__ __align__(16) _Float16 As[2][64 * LDSK];
  __shared__ __align__(16) _Float16 Bs[2][128 * LDSK];

  const int tid  = threadIdx.x;
  const int lane = tid & 31;
  const int waveId = tid >> 5;
  const int waveM = waveId & 3;          // 4 waves along rows (16 each)
  const int waveN = waveId >> 2;         // 2 waves along cols (64 each)
  const int rowBase = blockIdx.y * 64;   // batch rows
  const int colBase = blockIdx.x * 128;  // M cols

  const int lrow = lane & 15;
  const int koff = (lane & 16) ? 8 : 0;  // f16 A/B fragment K split per half-wave
  const int roff = (lane & 16) ? 8 : 0;  // C-matrix row split per half-wave

  // stage coords
  const int ar = tid >> 2, akc = (tid & 3) * 8;   // A: 64 rows x 32 k (8 f32 each)
  const int bj = tid >> 1, bkc = (tid & 1) * 16;  // B: 128 rows x 32 k (16 halves)

  v8f acc[4] = {};

  auto stage = [&](int buf, int k0) {
    // ---- B tile (f16 copy, async) ----
    const _Float16* bp = W1t + (size_t)(colBase + bj) * K1 + k0 + bkc;
    copy16B_to_lds(&Bs[buf][bj * LDSK + bkc],     bp);
    copy16B_to_lds(&Bs[buf][bj * LDSK + bkc + 8], bp + 8);
    // ---- A tile (f32 -> f16 convert through VGPRs) ----
    int kg = k0 + akc;
    const float* p = (kg < RNN_D)
        ? (x     + (size_t)(rowBase + ar) * RNN_D + kg)
        : (hprev + (size_t)(rowBase + ar) * RNN_H + (kg - RNN_D));
    float4 f0 = *(const float4*)p;
    float4 f1 = *(const float4*)(p + 4);
    *(v8h*)&As[buf][ar * LDSK + akc] = cvt8(f0, f1);
  };

  auto compute = [&](int cur) {
    const _Float16* ap = &As[cur][(waveM * 16 + lrow) * LDSK + koff];
    v16h afrag = cat8(*(const v8h*)ap, *(const v8h*)(ap + 16));
#pragma unroll
    for (int tile = 0; tile < 4; ++tile) {
      const _Float16* bp = &Bs[cur][(waveN * 64 + tile * 16 + lrow) * LDSK + koff];
      v16h bfrag = cat8(*(const v8h*)bp, *(const v8h*)(bp + 16));
      acc[tile] = __builtin_amdgcn_wmma_f32_16x16x32_f16(
          false, afrag, false, bfrag, (short)0, acc[tile], false, false);
    }
  };

  stage(0, 0);
  async_wait0();
  __syncthreads();

  const int NIT = K1 / 32;   // 32 (even)
  // Manual x2-unrolled pipeline, fixed buffer indices, no mid-loop exits.
#pragma clang loop unroll(disable)
  for (int k = 0; k + 2 < NIT; k += 2) {
    stage(1, (k + 1) * 32);
    compute(0);
    async_wait0();
    __syncthreads();
    stage(0, (k + 2) * 32);
    compute(1);
    async_wait0();
    __syncthreads();
  }
  // drain: steps NIT-2 (buf0, already staged) and NIT-1 (buf1)
  stage(1, (NIT - 1) * 32);
  compute(0);
  async_wait0();
  __syncthreads();
  compute(1);

  // ---- epilogue: + b1, tanh, store f16 ----
#pragma unroll
  for (int tile = 0; tile < 4; ++tile) {
    int col = colBase + waveN * 64 + tile * 16 + lrow;
    float bias = b1[col];
#pragma unroll
    for (int r = 0; r < 8; ++r) {
      int row = rowBase + waveM * 16 + r + roff;
      float v = tanhf(acc[tile][r] + bias);
      act[(size_t)row * RNN_M + col] = (_Float16)v;
    }
  }
}

// ---------------------------------------------------------------------------
// GEMM2: out_t[B, H] = act @ W2 + b2   (A already f16 -> fully async staging)
// ---------------------------------------------------------------------------
__global__ __launch_bounds__(256, 1) void rnn_gemm2(
    const _Float16* __restrict__ act, const _Float16* __restrict__ W2t,
    const float* __restrict__ b2, float* __restrict__ out_t) {
  __shared__ __align__(16) _Float16 As[2][64 * LDSK];
  __shared__ __align__(16) _Float16 Bs[2][128 * LDSK];

  const int tid  = threadIdx.x;
  const int lane = tid & 31;
  const int waveId = tid >> 5;
  const int waveM = waveId & 3;
  const int waveN = waveId >> 2;
  const int rowBase = blockIdx.y * 64;   // batch rows
  const int colBase = blockIdx.x * 128;  // H cols

  const int lrow = lane & 15;
  const int koff = (lane & 16) ? 8 : 0;
  const int roff = (lane & 16) ? 8 : 0;

  const int ar = tid >> 2, akc = (tid & 3) * 8;   // A: 64 x 32 halves
  const int bj = tid >> 1, bkc = (tid & 1) * 16;  // B: 128 x 32 halves

  v8f acc[4] = {};

  auto stage = [&](int buf, int k0) {
    const _Float16* apg = act + (size_t)(rowBase + ar) * RNN_M + k0 + akc;
    copy16B_to_lds(&As[buf][ar * LDSK + akc], apg);
    const _Float16* bpg = W2t + (size_t)(colBase + bj) * K2 + k0 + bkc;
    copy16B_to_lds(&Bs[buf][bj * LDSK + bkc],     bpg);
    copy16B_to_lds(&Bs[buf][bj * LDSK + bkc + 8], bpg + 8);
  };

  auto compute = [&](int cur) {
    const _Float16* ap = &As[cur][(waveM * 16 + lrow) * LDSK + koff];
    v16h afrag = cat8(*(const v8h*)ap, *(const v8h*)(ap + 16));
#pragma unroll
    for (int tile = 0; tile < 4; ++tile) {
      const _Float16* bp = &Bs[cur][(waveN * 64 + tile * 16 + lrow) * LDSK + koff];
      v16h bfrag = cat8(*(const v8h*)bp, *(const v8h*)(bp + 16));
      acc[tile] = __builtin_amdgcn_wmma_f32_16x16x32_f16(
          false, afrag, false, bfrag, (short)0, acc[tile], false, false);
    }
  };

  stage(0, 0);
  async_wait0();
  __syncthreads();

  const int NIT = K2 / 32;   // 64 (even)
#pragma clang loop unroll(disable)
  for (int k = 0; k + 2 < NIT; k += 2) {
    stage(1, (k + 1) * 32);
    compute(0);
    async_wait0();
    __syncthreads();
    stage(0, (k + 2) * 32);
    compute(1);
    async_wait0();
    __syncthreads();
  }
  stage(1, (NIT - 1) * 32);
  compute(0);
  async_wait0();
  __syncthreads();
  compute(1);

#pragma unroll
  for (int tile = 0; tile < 4; ++tile) {
    int col = colBase + waveN * 64 + tile * 16 + lrow;
    float bias = b2[col];
#pragma unroll
    for (int r = 0; r < 8; ++r) {
      int row = rowBase + waveM * 16 + r + roff;
      out_t[(size_t)row * RNN_H + col] = acc[tile][r] + bias;
    }
  }
}

// ---------------------------------------------------------------------------
extern "C" void kernel_launch(void* const* d_in, const int* in_sizes, int n_in,
                              void* d_out, int out_size, void* d_ws, size_t ws_size,
                              hipStream_t stream) {
  const float* x  = (const float*)d_in[0];   // [L, B, D]
  const float* h0 = (const float*)d_in[1];   // [1, B, H]
  const float* W1 = (const float*)d_in[2];   // [D+H, M]
  const float* b1 = (const float*)d_in[3];   // [M]
  const float* W2 = (const float*)d_in[4];   // [M, H]
  const float* b2 = (const float*)d_in[5];   // [H]
  float* out = (float*)d_out;                // [L,B,H] then h_last [1,B,H]

  _Float16* W1t = (_Float16*)d_ws;                       // [M,  K1] 4 MB
  _Float16* W2t = W1t + (size_t)RNN_M * K1;              // [H,  K2] 2 MB
  _Float16* act = W2t + (size_t)RNN_H * K2;              // [B,  M ] 1 MB

  const int nconv = RNN_M * K1 + RNN_H * K2;             // 3,145,728
  rnn_convert_weights<<<nconv / 256, 256, 0, stream>>>(W1, W2, W1t, W2t);

  for (int t = 0; t < RNN_L; ++t) {
    const float* xt = x + (size_t)t * RNN_B * RNN_D;
    const float* hp = (t == 0) ? h0 : out + (size_t)(t - 1) * RNN_B * RNN_H;
    float* ot = out + (size_t)t * RNN_B * RNN_H;

    rnn_gemm1_tanh<<<dim3(RNN_M / 128, RNN_B / 64), 256, 0, stream>>>(
        xt, hp, W1t, b1, act);
    rnn_gemm2<<<dim3(RNN_H / 128, RNN_B / 64), 256, 0, stream>>>(
        act, W2t, b2, ot);
  }

  // h_last = out[L-1]
  (void)hipMemcpyAsync(out + (size_t)RNN_L * RNN_B * RNN_H,
                       out + (size_t)(RNN_L - 1) * RNN_B * RNN_H,
                       (size_t)RNN_B * RNN_H * sizeof(float),
                       hipMemcpyDeviceToDevice, stream);
}
